// Next_Node_Probability_Calculator_for_group_39539468927611
// MI455X (gfx1250) — matrix operand order repaired and checked by reading, not compiled, not run on hardware
//
#include <hip/hip_runtime.h>
#include <hip/hip_bf16.h>
#include <math.h>

// ---------------------------------------------------------------------------
// MI455X (gfx1250): bf16 WMMA for all GEMM-shaped work, flash-attention
// accumulation with 32-key chunks (full-K PV WMMA), async global->LDS mask
// staging (ASYNCcnt), 320KB-WGP-LDS resident score buffer, one coalesced
// output pass.  wave32 throughout.
// ---------------------------------------------------------------------------

typedef __attribute__((ext_vector_type(16))) __bf16 v16bf;
typedef __attribute__((ext_vector_type(8)))  float  v8f;

#define EMB   128
#define HK    128     // H*K
#define NH    8
#define KD    16
#define B_    64
#define G_    512
#define P_    2000
#define PPAD  2016    // padded to multiple of 32
#define GT    16      // query-tile rows per block
#define CLIPV 10.0f

__device__ __forceinline__ __bf16 f2bf(float x) { return (__bf16)x; }

__device__ __forceinline__ v8f wmma_bf16(v16bf a, v16bf b, v8f c) {
  // D = A(16x32 bf16) x B(32x16 bf16) + C(16x16 f32)
  return __builtin_amdgcn_wmma_f32_16x16x32_bf16(
      false, a, false, b, (short)0, c, false, false);
}

// ---------------------------------------------------------------------------
// Kernel 1: K/V projections (encoded @ Wk, encoded @ Wv) -> bf16, P padded.
// ---------------------------------------------------------------------------
__global__ void kv_project_kernel(const float* __restrict__ enc,
                                  const float* __restrict__ Wk,
                                  const float* __restrict__ Wv,
                                  __bf16* __restrict__ kproj,
                                  __bf16* __restrict__ vproj) {
  int lane = threadIdx.x & 31;
  int wave = threadIdx.x >> 5;
  int tile = blockIdx.x * 4 + wave;
  int row0 = tile * 16;
  int b  = row0 / PPAD;
  int p0 = row0 - b * PPAD;
  int lo = lane & 15, hi = lane >> 4;

  if (p0 >= P_) {              // pad rows -> zeros (whole wave uniform)
    #pragma unroll
    for (int r = 0; r < 8; ++r) {
      long ro = (long)(b * PPAD + p0 + r + hi * 8) * HK + lo;
      for (int nt = 0; nt < 8; ++nt) {
        kproj[ro + nt * 16] = (__bf16)0.0f;
        vproj[ro + nt * 16] = (__bf16)0.0f;
      }
    }
    return;
  }

  const float* arow = enc + (long)(b * P_ + p0 + lo) * EMB;
  v16bf a[4];
  #pragma unroll
  for (int kb = 0; kb < 4; ++kb) {
    #pragma unroll
    for (int j = 0; j < 8; ++j) {
      int kk = kb * 32 + ((j >> 2) * 16) + hi * 8 + (j & 3) * 2;
      a[kb][2 * j]     = f2bf(arow[kk]);
      a[kb][2 * j + 1] = f2bf(arow[kk + 1]);
    }
  }
  for (int w = 0; w < 2; ++w) {
    const float* W = w ? Wv : Wk;
    __bf16* outp   = w ? vproj : kproj;
    for (int nt = 0; nt < 8; ++nt) {
      int n = nt * 16 + lo;
      v8f acc = {0.f,0.f,0.f,0.f,0.f,0.f,0.f,0.f};
      #pragma unroll
      for (int kb = 0; kb < 4; ++kb) {
        v16bf bf;
        #pragma unroll
        for (int j = 0; j < 8; ++j) {
          int kk = kb * 32 + ((j >> 2) * 16) + hi * 8 + (j & 3) * 2;
          bf[2 * j]     = f2bf(W[(long)kk * HK + n]);
          bf[2 * j + 1] = f2bf(W[(long)(kk + 1) * HK + n]);
        }
        acc = wmma_bf16(a[kb], bf, acc);
      }
      #pragma unroll
      for (int r = 0; r < 8; ++r)
        outp[(long)(b * PPAD + p0 + r + hi * 8) * HK + n] = f2bf(acc[r]);
    }
  }
}

// ---------------------------------------------------------------------------
// Kernel 2: encoded_nodes -> bf16 (padded) for the second-score GEMM.
// ---------------------------------------------------------------------------
__global__ void en_convert_kernel(const float* __restrict__ enc,
                                  __bf16* __restrict__ en_bf) {
  long idx = (long)blockIdx.x * blockDim.x + threadIdx.x;
  long total = (long)B_ * PPAD * EMB;
  if (idx >= total) return;
  int e = (int)(idx % EMB);
  long row = idx / EMB;
  int b = (int)(row / PPAD);
  int p = (int)(row % PPAD);
  float v = (p < P_) ? enc[(long)(b * P_ + p) * EMB + e] : 0.0f;
  en_bf[idx] = f2bf(v);
}

// ---------------------------------------------------------------------------
// Kernel 3a/3b: Q projection via rank-1 structure:
//   q[b,g,:] = graph[b]@Wq[:128,:] + cap[b,g]*Wq[128,:];  scaled by 1/sqrt(16)
// ---------------------------------------------------------------------------
__global__ void qbase_kernel(const float* __restrict__ graph,
                             const float* __restrict__ Wq,
                             float* __restrict__ qbase) {
  int b = blockIdx.x, n = threadIdx.x;
  float acc = 0.f;
  for (int e = 0; e < EMB; ++e) acc += graph[b * EMB + e] * Wq[e * HK + n];
  qbase[b * HK + n] = acc;
}

__global__ void qproj_kernel(const float* __restrict__ capacity,
                             const float* __restrict__ Wq,
                             const float* __restrict__ qbase,
                             __bf16* __restrict__ qbf) {
  long idx = (long)blockIdx.x * blockDim.x + threadIdx.x;
  if (idx >= (long)B_ * G_ * HK) return;
  int n = (int)(idx % HK);
  long bg = idx / HK;
  int b = (int)(bg / G_);
  float q = qbase[b * HK + n] + capacity[bg] * Wq[EMB * HK + n];
  qbf[idx] = f2bf(q * 0.25f);   // 1/sqrt(K)
}

// ---------------------------------------------------------------------------
// Kernel 4: top-100-smallest bonus via rank counting (JAX tie-break: lower
// index wins).  rank 1..99  -> -log(r);  else -> -r.  Pad -> 0.
// ---------------------------------------------------------------------------
__global__ void bonus_kernel(const float* __restrict__ ratios,
                             float* __restrict__ bonus) {
  __shared__ float rs[P_];
  int b = blockIdx.x;
  for (int p = threadIdx.x; p < P_; p += blockDim.x) rs[p] = ratios[b * P_ + p];
  __syncthreads();
  for (int p = threadIdx.x; p < PPAD; p += blockDim.x) {
    float out;
    if (p >= P_) {
      out = 0.0f;
    } else {
      float rp = rs[p];
      int cnt = 0;
      for (int p2 = 0; p2 < P_; ++p2) {
        float r2 = rs[p2];
        cnt += (r2 < rp) || (r2 == rp && p2 < p);
      }
      out = (cnt >= 1 && cnt <= 99) ? -logf(rp) : -rp;
    }
    bonus[b * PPAD + p] = out;
  }
}

// ---------------------------------------------------------------------------
// Kernel 5: fused per-(b, 16-query-tile) block: 8 waves, wave h = head h.
// Phase 1: flash attention per head, 32-key chunks (2 QK^T WMMA + 1 full-K
//          PV WMMA).  Mask tile staged to LDS with async global->LDS loads.
// Phase 2: out@Wc+bc, second score vs encoded^T (WMMA), bonus/tanh/mask,
//          block softmax over P from the LDS score buffer, coalesced store.
// ---------------------------------------------------------------------------
__launch_bounds__(256, 1)
__global__ void attention_kernel(const float* __restrict__ ninf,
                                 const float* __restrict__ Wc,
                                 const float* __restrict__ bcv,
                                 const __bf16* __restrict__ kproj,
                                 const __bf16* __restrict__ vproj,
                                 const __bf16* __restrict__ en_bf,
                                 const __bf16* __restrict__ qbf,
                                 const float* __restrict__ bonus,
                                 float* __restrict__ out) {
  extern __shared__ char smem[];
  float*  mask_s  = (float*)smem;                          // 16*PPAD f32
  float*  score_s = mask_s + 16 * PPAD;                    // 16*PPAD f32
  __bf16* oc_s    = (__bf16*)(score_s + 16 * PPAD);        // 16*128 bf16
  __bf16* mh_s    = oc_s + 16 * HK;                        // 16*128 bf16
  __bf16* pt_s    = mh_s + 16 * HK;                        // 8 * 16*32 bf16
  float*  red_s   = (float*)(pt_s + 8 * 512);              // 16*16 f32
  float*  row_s   = red_s + 16 * 16;                       // 32 f32

  int tid  = threadIdx.x;
  int lane = tid & 31;
  int wave = tid >> 5;              // == head in phase 1
  int lo = lane & 15, hi = lane >> 4;
  int bidx  = blockIdx.x >> 5;      // G_/GT = 32 tiles/batch
  int gbase = (blockIdx.x & 31) * GT;
  long kvbase = (long)bidx * PPAD * HK;

  // ---- mask tile -> LDS via async global->LDS (B128), pad cols -1e30 ----
  {
    const float* gptr = ninf + (long)(bidx * G_ + gbase) * P_;
    unsigned mask_lds = (unsigned)(size_t)(void*)mask_s;
    for (int t = tid; t < 16 * (P_ / 4); t += 256) {       // 16 rows * 500 chunks
      int row = t / (P_ / 4), cch = t - row * (P_ / 4);
      unsigned lds = mask_lds + (unsigned)(row * PPAD + cch * 4) * 4u;
      unsigned long long ga =
          (unsigned long long)(size_t)(gptr + (long)row * P_ + cch * 4);
      asm volatile("global_load_async_to_lds_b128 %0, %1, off"
                   :: "v"(lds), "v"(ga) : "memory");
    }
    for (int t = tid; t < 16 * (PPAD - P_); t += 256) {    // 256 pad elements
      int row = t / (PPAD - P_), p = P_ + (t - row * (PPAD - P_));
      mask_s[row * PPAD + p] = -1e30f;
    }
    asm volatile("s_wait_asynccnt 0" ::: "memory");
  }
  __syncthreads();

  // ---- Phase 1: flash attention, head h = wave, 32-key chunks ----
  int h = wave;
  v16bf qa;   // A-frag: M=query, K(real)=0..15 head dims, K=16..31 zero-pad
  {
    const __bf16* qrow = qbf + (long)(bidx * G_ + gbase + lo) * HK + h * KD;
    #pragma unroll
    for (int j = 0; j < 8; ++j) {
      int kk = 2 * j;
      qa[2 * j]     = hi ? (__bf16)0.f : qrow[kk];
      qa[2 * j + 1] = hi ? (__bf16)0.f : qrow[kk + 1];
    }
  }
  float m[8], l[8];
  v8f acc = {0.f,0.f,0.f,0.f,0.f,0.f,0.f,0.f};
  #pragma unroll
  for (int r = 0; r < 8; ++r) { m[r] = -1e30f; l[r] = 0.f; }
  __bf16* pt = pt_s + wave * 512;   // 16 rows x 32 keys
  v8f zc = {0.f,0.f,0.f,0.f,0.f,0.f,0.f,0.f};

  for (int c = 0; c < PPAD / 32; ++c) {                    // 63 chunks
    int pbase = c * 32;
    // two K B-frags (16 keys each): N=key, K(real)=head dims 0..15
    v16bf kf0, kf1;
    if (hi == 0) {
      const __bf16* kp0 = kproj + kvbase + (long)(pbase + lo) * HK + h * KD;
      const __bf16* kp1 = kp0 + 16 * HK;
      __builtin_prefetch(kp0 + 32 * HK, 0, 1);
      #pragma unroll
      for (int j = 0; j < 8; ++j) {
        kf0[2*j] = kp0[2*j]; kf0[2*j+1] = kp0[2*j+1];
        kf1[2*j] = kp1[2*j]; kf1[2*j+1] = kp1[2*j+1];
      }
    } else {
      #pragma unroll
      for (int j = 0; j < 8; ++j) {
        kf0[2*j] = (__bf16)0.f; kf0[2*j+1] = (__bf16)0.f;
        kf1[2*j] = (__bf16)0.f; kf1[2*j+1] = (__bf16)0.f;
      }
    }
    v8f s0 = wmma_bf16(qa, kf0, zc);
    v8f s1 = wmma_bf16(qa, kf1, zc);

    #pragma unroll
    for (int r = 0; r < 8; ++r) {
      int row = r + hi * 8;
      float sv0 = s0[r] + mask_s[row * PPAD + pbase + lo];
      float sv1 = s1[r] + mask_s[row * PPAD + pbase + 16 + lo];
      float cm = fmaxf(sv0, sv1);
      cm = fmaxf(cm, __shfl_xor(cm, 1, 32));
      cm = fmaxf(cm, __shfl_xor(cm, 2, 32));
      cm = fmaxf(cm, __shfl_xor(cm, 4, 32));
      cm = fmaxf(cm, __shfl_xor(cm, 8, 32));
      float mn = fmaxf(m[r], cm);
      float sc = __expf(m[r] - mn);
      float pe0 = __expf(sv0 - mn);
      float pe1 = __expf(sv1 - mn);
      float ps = pe0 + pe1;
      ps += __shfl_xor(ps, 1, 32);
      ps += __shfl_xor(ps, 2, 32);
      ps += __shfl_xor(ps, 4, 32);
      ps += __shfl_xor(ps, 8, 32);
      l[r] = l[r] * sc + ps;
      m[r] = mn;
      acc[r] *= sc;
      pt[row * 32 + lo]      = f2bf(pe0);
      pt[row * 32 + 16 + lo] = f2bf(pe1);
    }
    asm volatile("s_wait_dscnt 0" ::: "memory");
    // probs A-frag: M=query, K(real)=32 keys -> full-K PV WMMA
    v16bf pa;
    {
      const __bf16* pr = pt + lo * 32;
      #pragma unroll
      for (int j = 0; j < 8; ++j) {
        int kk = ((j >> 2) * 16) + hi * 8 + (j & 3) * 2;
        pa[2 * j]     = pr[kk];
        pa[2 * j + 1] = pr[kk + 1];
      }
    }
    // V B-frag: N=head dim, K=32 keys (both lane halves live)
    v16bf vf;
    {
      const __bf16* vp = vproj + kvbase + (long)(pbase + hi * 8) * HK + h * KD + lo;
      #pragma unroll
      for (int j = 0; j < 8; ++j) {
        vf[2 * j]     = vp[(2 * j + (j >> 2) * 8) * HK];
        vf[2 * j + 1] = vp[(2 * j + 1 + (j >> 2) * 8) * HK];
      }
    }
    acc = wmma_bf16(pa, vf, acc);
  }
  #pragma unroll
  for (int r = 0; r < 8; ++r)
    oc_s[(r + hi * 8) * HK + h * KD + lo] = f2bf(acc[r] / l[r]);
  __syncthreads();

  // ---- Phase 2a: mh = out_concat @ Wc + bc; wave w -> 16-col tile ----
  {
    int n = wave * 16 + lo;
    v8f a2 = {0.f,0.f,0.f,0.f,0.f,0.f,0.f,0.f};
    #pragma unroll
    for (int kb = 0; kb < 4; ++kb) {
      v16bf af;
      const __bf16* ar = oc_s + lo * HK + kb * 32;
      #pragma unroll
      for (int j = 0; j < 8; ++j) {
        int kk = ((j >> 2) * 16) + hi * 8 + (j & 3) * 2;
        af[2 * j] = ar[kk]; af[2 * j + 1] = ar[kk + 1];
      }
      v16bf bf;
      #pragma unroll
      for (int j = 0; j < 8; ++j) {
        int kk = kb * 32 + ((j >> 2) * 16) + hi * 8 + (j & 3) * 2;
        bf[2 * j]     = f2bf(Wc[kk * HK + n]);
        bf[2 * j + 1] = f2bf(Wc[(kk + 1) * HK + n]);
      }
      a2 = wmma_bf16(af, bf, a2);
    }
    float bias = bcv[n];
    #pragma unroll
    for (int r = 0; r < 8; ++r)
      mh_s[(r + hi * 8) * HK + n] = f2bf(a2[r] + bias);
  }
  __syncthreads();

  // ---- Phase 2b: score2 = mh @ en^T / sqrt(128) + bonus; tanh; +mask ----
  const float inv_se = 0.08838834764831845f;   // 1/sqrt(128)
  v16bf mha[4];
  #pragma unroll
  for (int kb = 0; kb < 4; ++kb) {
    const __bf16* ar = mh_s + lo * HK + kb * 32;
    #pragma unroll
    for (int j = 0; j < 8; ++j) {
      int kk = ((j >> 2) * 16) + hi * 8 + (j & 3) * 2;
      mha[kb][2 * j] = ar[kk]; mha[kb][2 * j + 1] = ar[kk + 1];
    }
  }
  for (int c = wave; c < PPAD / 16; c += 8) {
    int pbase = c * 16;
    __builtin_prefetch(en_bf + kvbase + (long)(pbase + 128 + lo) * HK, 0, 1);
    v8f a3 = {0.f,0.f,0.f,0.f,0.f,0.f,0.f,0.f};
    #pragma unroll
    for (int kb = 0; kb < 4; ++kb) {
      v16bf bf;
      const __bf16* er = en_bf + kvbase + (long)(pbase + lo) * HK + kb * 32 + hi * 8;
      #pragma unroll
      for (int j = 0; j < 8; ++j) { int kk = ((j >> 2) * 16) + (j & 3) * 2; bf[2*j] = er[kk]; bf[2*j+1] = er[kk+1]; }
      a3 = wmma_bf16(mha[kb], bf, a3);
    }
    int p = pbase + lo;
    float bn = bonus[bidx * PPAD + p];
    #pragma unroll
    for (int r = 0; r < 8; ++r) {
      int row = r + hi * 8;
      float v = a3[r] * inv_se + bn;
      v = CLIPV * tanhf(v);
      v += mask_s[row * PPAD + p];
      score_s[row * PPAD + p] = v;
    }
  }
  __syncthreads();

  // ---- Phase 2c: softmax over P per row, coalesced write ----
  int row = tid >> 4;
  int sl  = tid & 15;
  float pm = -1e30f;
  for (int p = sl; p < PPAD; p += 16) pm = fmaxf(pm, score_s[row * PPAD + p]);
  red_s[row * 16 + sl] = pm;
  __syncthreads();
  if (sl == 0) {
    float mm = -1e30f;
    for (int i = 0; i < 16; ++i) mm = fmaxf(mm, red_s[row * 16 + i]);
    row_s[row] = mm;
  }
  __syncthreads();
  float rm = row_s[row];
  float psum = 0.f;
  for (int p = sl; p < PPAD; p += 16) psum += __expf(score_s[row * PPAD + p] - rm);
  red_s[row * 16 + sl] = psum;
  __syncthreads();
  if (sl == 0) {
    float ss = 0.f;
    for (int i = 0; i < 16; ++i) ss += red_s[row * 16 + i];
    row_s[16 + row] = 1.0f / ss;
  }
  __syncthreads();
  float inv = row_s[16 + row];
  float* orow = out + (long)(bidx * G_ + gbase + row) * P_;
  for (int p = sl; p < P_; p += 16)
    orow[p] = __expf(score_s[row * PPAD + p] - rm) * inv;
}

// ---------------------------------------------------------------------------
extern "C" void kernel_launch(void* const* d_in, const int* in_sizes, int n_in,
                              void* d_out, int out_size, void* d_ws, size_t ws_size,
                              hipStream_t stream) {
  const float* graph    = (const float*)d_in[0];
  const float* capacity = (const float*)d_in[1];
  const float* ninf     = (const float*)d_in[2];
  const float* ratios   = (const float*)d_in[3];
  const float* enc      = (const float*)d_in[4];
  const float* Wq       = (const float*)d_in[5];
  const float* Wk       = (const float*)d_in[6];
  const float* Wv       = (const float*)d_in[7];
  const float* Wc       = (const float*)d_in[8];
  const float* bcv      = (const float*)d_in[9];

  char* ws = (char*)d_ws;
  size_t off = 0;
  auto take = [&](size_t bytes) { char* p = ws + off; off = (off + bytes + 255) & ~(size_t)255; return p; };
  __bf16* kproj = (__bf16*)take((size_t)B_ * PPAD * HK * 2);
  __bf16* vproj = (__bf16*)take((size_t)B_ * PPAD * HK * 2);
  __bf16* en_bf = (__bf16*)take((size_t)B_ * PPAD * EMB * 2);
  __bf16* qbf   = (__bf16*)take((size_t)B_ * G_ * HK * 2);
  float*  qbase = (float*) take((size_t)B_ * HK * 4);
  float*  bonus = (float*) take((size_t)B_ * PPAD * 4);

  kv_project_kernel<<<(B_ * PPAD / 16) / 4, 128, 0, stream>>>(enc, Wk, Wv, kproj, vproj);
  {
    long total = (long)B_ * PPAD * EMB;
    en_convert_kernel<<<(int)((total + 255) / 256), 256, 0, stream>>>(enc, en_bf);
  }
  qbase_kernel<<<B_, 128, 0, stream>>>(graph, Wq, qbase);
  {
    long total = (long)B_ * G_ * HK;
    qproj_kernel<<<(int)((total + 255) / 256), 256, 0, stream>>>(capacity, Wq, qbase, qbf);
  }
  bonus_kernel<<<B_, 256, 0, stream>>>(ratios, bonus);

  size_t smem = (size_t)(16 * PPAD) * 4 * 2          // mask + score (f32)
              + (size_t)(16 * HK) * 2 * 2            // oc + mh (bf16)
              + (size_t)(8 * 512) * 2                // per-wave transpose (bf16)
              + (size_t)(16 * 16) * 4 + 32 * 4;      // reductions
  attention_kernel<<<B_ * (G_ / GT), 256, smem, stream>>>(
      ninf, Wc, bcv, kproj, vproj, en_bf, qbf, bonus, (float*)d_out);
}
